// MultiHeadSelfAttention_25340307046929
// MI455X (gfx1250) — compile-verified, matrix operand order checked by reading, not run on hardware
//
#include <hip/hip_runtime.h>
#include <math.h>

typedef __attribute__((ext_vector_type(16))) _Float16 v16h;
typedef __attribute__((ext_vector_type(8)))  float    v8f;
using half_t = _Float16;

#define B_   2
#define S_   2048
#define D_   1024
#define H_   16
#define DK_  64
#define MTOT (B_*S_)      // 4096

union FragH { v16h h; uint4 u[2]; };

// CDNA5 16-bit A/B fragment (16 rows x 32 K): lane holds row (base+lane%16);
// halves 0..7 -> K = k0 + group*8 + {0..7}; halves 8..15 -> K = k0+16+group*8+{0..7}
__device__ inline v16h load_frag(const half_t* __restrict__ base, int ld,
                                 int row, int k0, int lane) {
    FragH f;
    const half_t* p = base + (size_t)(row + (lane & 15)) * ld + k0 + ((lane >> 4) << 3);
    f.u[0] = *(const uint4*)(p);
    f.u[1] = *(const uint4*)(p + 16);
    return f.h;
}

// ---------------- fp32 -> fp16 conversion ----------------
__global__ void f32_to_f16(const float* __restrict__ s, half_t* __restrict__ d, int n) {
    int i = blockIdx.x * blockDim.x + threadIdx.x;
    if (i < n) d[i] = (half_t)s[i];
}

// ---------------- WMMA GEMM: C[M][N] = A[M][K] * B[N][K]^T (f16 in, f32 out) ----
// grid = (N/64, M/256), block = 256 (8 waves); wave computes 32(M) x 64(N):
// per 32-K step: 4 A-loads + 8 B-loads feed 8 WMMAs (B fragments reused 2x).
__global__ __launch_bounds__(256) void gemm_f16_f32(
    const half_t* __restrict__ A, const half_t* __restrict__ Bm,
    float* __restrict__ C, int M, int N, int K)
{
    const int lane = threadIdx.x & 31;
    const int wave = threadIdx.x >> 5;
    const int m0 = blockIdx.y * 256 + wave * 32;
    const int n0 = blockIdx.x * 64;

    v8f z = {0.f,0.f,0.f,0.f,0.f,0.f,0.f,0.f};
    v8f acc0[4], acc1[4];
    #pragma unroll
    for (int c = 0; c < 4; ++c) { acc0[c] = z; acc1[c] = z; }

    const half_t* aPre = A + (size_t)(m0 + (lane & 15)) * K;
    const half_t* bPre = Bm + (size_t)(n0 + (lane & 15)) * K;

    for (int k0 = 0; k0 < K; k0 += 32) {
        // prefetch next k-tile of the streaming A rows and the L2-resident B rows
        if (k0 + 64 < K) {
            __builtin_prefetch(aPre + k0 + 64, 0, 1);
            __builtin_prefetch(bPre + k0 + 64, 0, 1);
        }
        v16h a0 = load_frag(A, K, m0,      k0, lane);
        v16h a1 = load_frag(A, K, m0 + 16, k0, lane);
        #pragma unroll
        for (int c = 0; c < 4; ++c) {
            v16h b = load_frag(Bm, K, n0 + c * 16, k0, lane);
            acc0[c] = __builtin_amdgcn_wmma_f32_16x16x32_f16(
                false, a0, false, b, (short)0, acc0[c], false, false);
            acc1[c] = __builtin_amdgcn_wmma_f32_16x16x32_f16(
                false, a1, false, b, (short)0, acc1[c], false, false);
        }
    }
    // f32 C/D layout: lane = column n0+lane%16; VGPR r = row m0 + r + 8*(lane/16)
    const int g = lane >> 4, col = lane & 15;
    #pragma unroll
    for (int c = 0; c < 4; ++c)
        #pragma unroll
        for (int r = 0; r < 8; ++r) {
            C[(size_t)(m0 + r + 8 * g) * N + n0 + c * 16 + col]      = acc0[c][r];
            C[(size_t)(m0 + 16 + r + 8 * g) * N + n0 + c * 16 + col] = acc1[c][r];
        }
}

// ---------------- RoPE + head split + V transpose ----------------
// QKV f32 [4096][3072] -> Qh,Kh f16 [b,h,s,64] and Vt f16 [b,h,64,s]
__global__ void rope_split(const float* __restrict__ QKV, const int* __restrict__ pos,
                           const int* __restrict__ thetaP,
                           half_t* __restrict__ Qh, half_t* __restrict__ Kh,
                           half_t* __restrict__ Vt)
{
    int i = blockIdx.x * blockDim.x + threadIdx.x;    // B*S*H*32 = 2^21 threads
    if (i >= B_ * S_ * H_ * (DK_ / 2)) return;
    const float theta = (float)(*thetaP);
    int pr = i & 31;
    int h  = (i >> 5) & 15;
    int s  = (i >> 9) & 2047;
    int b  = i >> 20;

    size_t row  = (size_t)(b * S_ + s) * 3072;
    size_t qoff = row + h * DK_;
    int d0 = 2 * pr;

    float p   = (float)pos[s];
    float inv = __powf(theta, -(float)pr / 32.0f);   // theta^(-2*pr/64)
    float ang = p * inv;
    float sn = __sinf(ang), cs = __cosf(ang);

    float qe = QKV[qoff + d0],        qo = QKV[qoff + d0 + 1];
    float ke = QKV[qoff + 1024 + d0], ko = QKV[qoff + 1024 + d0 + 1];
    float ve = QKV[qoff + 2048 + d0], vo = QKV[qoff + 2048 + d0 + 1];

    size_t bh   = (size_t)(b * H_ + h);
    size_t qdst = (bh * S_ + s) * DK_ + d0;
    Qh[qdst]     = (half_t)(qe * cs - qo * sn);
    Qh[qdst + 1] = (half_t)(qe * sn + qo * cs);
    Kh[qdst]     = (half_t)(ke * cs - ko * sn);
    Kh[qdst + 1] = (half_t)(ke * sn + ko * cs);

    size_t vdst = (bh * DK_ + d0) * S_ + s;
    Vt[vdst]      = (half_t)ve;
    Vt[vdst + S_] = (half_t)vo;
}

// ---------------- Flash attention: 1 wave per 16-query tile ----------------
// grid = B*H*(S/16) = 4096 blocks of 32 threads
__global__ __launch_bounds__(32) void flash_attn(
    const half_t* __restrict__ Qh, const half_t* __restrict__ Kh,
    const half_t* __restrict__ Vt, half_t* __restrict__ Oh)
{
    __shared__ __align__(16) half_t P[16 * 32];   // exp(scores) tile, A-layout staging

    const int lane = threadIdx.x & 31;
    const int qt = blockIdx.x & 127;
    const int h  = (blockIdx.x >> 7) & 15;
    const int b  = blockIdx.x >> 11;
    const int q0 = qt * 16;

    const size_t bh = (size_t)(b * H_ + h);
    const half_t* Qb = Qh + bh * S_ * DK_;
    const half_t* Kb = Kh + bh * S_ * DK_;
    const half_t* Vb = Vt + bh * DK_ * S_;

    const v16h qa0 = load_frag(Qb, DK_, q0, 0,  lane);
    const v16h qa1 = load_frag(Qb, DK_, q0, 32, lane);

    v8f z = {0.f,0.f,0.f,0.f,0.f,0.f,0.f,0.f};
    v8f o[4];
    float mi[8], li[8], alpha[8];
    #pragma unroll
    for (int c = 0; c < 4; ++c) o[c] = z;
    #pragma unroll
    for (int r = 0; r < 8; ++r) { mi[r] = -INFINITY; li[r] = 0.f; }

    const int g = lane >> 4, col = lane & 15;
    const float scale = 0.125f;   // 1/sqrt(64)

    for (int j = 0; j < q0 + 16; j += 32) {   // causal: only keys <= q0+15
        v16h kb00 = load_frag(Kb, DK_, j,      0,  lane);
        v16h kb01 = load_frag(Kb, DK_, j,      32, lane);
        v16h kb10 = load_frag(Kb, DK_, j + 16, 0,  lane);
        v16h kb11 = load_frag(Kb, DK_, j + 16, 32, lane);

        v8f s0 = __builtin_amdgcn_wmma_f32_16x16x32_f16(false, qa0, false, kb00, (short)0, z,  false, false);
        s0     = __builtin_amdgcn_wmma_f32_16x16x32_f16(false, qa1, false, kb01, (short)0, s0, false, false);
        v8f s1 = __builtin_amdgcn_wmma_f32_16x16x32_f16(false, qa0, false, kb10, (short)0, z,  false, false);
        s1     = __builtin_amdgcn_wmma_f32_16x16x32_f16(false, qa1, false, kb11, (short)0, s1, false, false);

        #pragma unroll
        for (int r = 0; r < 8; ++r) {
            const int q = q0 + r + 8 * g;
            float v0 = (j + col       > q) ? -1e9f : s0[r] * scale;
            float v1 = (j + 16 + col  > q) ? -1e9f : s1[r] * scale;
            // row-wise max across the 16 lanes holding this row
            float rm = fmaxf(v0, v1);
            rm = fmaxf(rm, __shfl_xor(rm, 1, 32));
            rm = fmaxf(rm, __shfl_xor(rm, 2, 32));
            rm = fmaxf(rm, __shfl_xor(rm, 4, 32));
            rm = fmaxf(rm, __shfl_xor(rm, 8, 32));
            float mnew = fmaxf(mi[r], rm);
            float al = __expf(mi[r] - mnew);
            float p0 = __expf(v0 - mnew);
            float p1 = __expf(v1 - mnew);
            float rs = p0 + p1;
            rs += __shfl_xor(rs, 1, 32);
            rs += __shfl_xor(rs, 2, 32);
            rs += __shfl_xor(rs, 4, 32);
            rs += __shfl_xor(rs, 8, 32);
            li[r] = li[r] * al + rs;
            mi[r] = mnew;
            alpha[r] = al;
            P[(r + 8 * g) * 32 + col]      = (half_t)p0;
            P[(r + 8 * g) * 32 + 16 + col] = (half_t)p1;
        }
        __builtin_amdgcn_wave_barrier();   // LDS ops are in-order within a wave

        #pragma unroll
        for (int c = 0; c < 4; ++c)
            #pragma unroll
            for (int r = 0; r < 8; ++r)
                o[c][r] *= alpha[r];

        // reload P as a 16x32 f16 A-fragment (ds_load_b128 x2 per lane)
        v16h pa = load_frag((const half_t*)P, 32, 0, 0, lane);

        #pragma unroll
        for (int c = 0; c < 4; ++c) {
            v16h vb = load_frag(Vb, S_, c * 16, j, lane);  // rows = dk, K = keys
            o[c] = __builtin_amdgcn_wmma_f32_16x16x32_f16(
                false, pa, false, vb, (short)0, o[c], false, false);
        }
        __builtin_amdgcn_wave_barrier();
    }

    #pragma unroll
    for (int c = 0; c < 4; ++c)
        #pragma unroll
        for (int r = 0; r < 8; ++r) {
            const int m = r + 8 * g;
            Oh[((size_t)(b * S_) + q0 + m) * D_ + h * DK_ + c * 16 + col] =
                (half_t)(o[c][r] / li[r]);
        }
}

extern "C" void kernel_launch(void* const* d_in, const int* in_sizes, int n_in,
                              void* d_out, int out_size, void* d_ws, size_t ws_size,
                              hipStream_t stream) {
    (void)in_sizes; (void)n_in; (void)out_size; (void)ws_size;
    const float* x   = (const float*)d_in[0];
    const float* Wq  = (const float*)d_in[1];
    const float* Wk  = (const float*)d_in[2];
    const float* Wv  = (const float*)d_in[3];
    const float* Wo  = (const float*)d_in[4];
    const int*   pos = (const int*)d_in[5];
    const int* theta = (const int*)d_in[7];

    char* ws = (char*)d_ws;
    half_t* xh    = (half_t*)(ws);                         //  8 MB: x f16 [4096][1024]
    half_t* Wqkvh = (half_t*)(ws + (size_t)( 8u << 20));   //  6 MB: [3072][1024]
    half_t* Woh   = (half_t*)(ws + (size_t)(14u << 20));   //  2 MB: [1024][1024]
    float*  QKV   = (float*) (ws + (size_t)(16u << 20));   // 48 MB: [4096][3072]
    half_t* Qh    = (half_t*)(ws + (size_t)(64u << 20));   //  8 MB: [b,h,s,64]
    half_t* Kh    = (half_t*)(ws + (size_t)(72u << 20));   //  8 MB
    half_t* Vt    = (half_t*)(ws + (size_t)(80u << 20));   //  8 MB: [b,h,64,s]
    half_t* Oh    = (half_t*)(ws + (size_t)(88u << 20));   //  8 MB: [4096][1024]

    const int DD = D_ * D_;                                // 1048576
    f32_to_f16<<<(MTOT * D_ + 255) / 256, 256, 0, stream>>>(x, xh, MTOT * D_);
    f32_to_f16<<<(DD + 255) / 256, 256, 0, stream>>>(Wq, Wqkvh,            DD);
    f32_to_f16<<<(DD + 255) / 256, 256, 0, stream>>>(Wk, Wqkvh + DD,       DD);
    f32_to_f16<<<(DD + 255) / 256, 256, 0, stream>>>(Wv, Wqkvh + 2 * DD,   DD);
    f32_to_f16<<<(DD + 255) / 256, 256, 0, stream>>>(Wo, Woh,              DD);

    // QKV = x * W_qkv^T   : M=4096, N=3072, K=1024
    gemm_f16_f32<<<dim3(3072 / 64, MTOT / 256), 256, 0, stream>>>(
        xh, Wqkvh, QKV, MTOT, 3072, D_);

    rope_split<<<(B_ * S_ * H_ * (DK_ / 2)) / 256, 256, 0, stream>>>(
        QKV, pos, theta, Qh, Kh, Vt);

    flash_attn<<<B_ * H_ * (S_ / 16), 32, 0, stream>>>(Qh, Kh, Vt, Oh);

    // out = O * W_o^T : M=4096, N=1024, K=1024
    gemm_f16_f32<<<dim3(D_ / 64, MTOT / 256), 256, 0, stream>>>(
        Oh, Woh, (float*)d_out, MTOT, D_, D_);
}
